// GeneralAttention_75746043232735
// MI455X (gfx1250) — matrix operand order repaired and checked by reading, not compiled, not run on hardware
//
#include <hip/hip_runtime.h>
#include <hip/hip_bf16.h>

// ---------------------------------------------------------------------------
// MI455X (gfx1250) attention: f16 WMMA GEMMs + f32 accumulate/softmax.
//   stage0: cast/pad/transpose prep
//   stage1: QKV GEMM   (WMMA f32_16x16x32_f16, async-LDS double buffered)
//   stage2: fused scores * Mij -> softmax -> PV (WMMA, K/V staged in LDS)
//   stage3: proj GEMM + bias (WMMA, async-LDS double buffered)
// ---------------------------------------------------------------------------

typedef __attribute__((ext_vector_type(16))) _Float16 v16h;
typedef __attribute__((ext_vector_type(8)))  float    v8f;
typedef int v4i __attribute__((vector_size(16)));

#define NQ    197          // real sequence length (14*14+1)
#define NP    224          // padded to 7*32 so all K loops are WMMA-K multiples
#define DIMC  768
#define HEADS 12
#define HD    64
#define BATCH 64

// ---- CDNA5 async global->LDS copy (ASYNCcnt path), with safe fallback -----
#if __has_builtin(__builtin_amdgcn_global_load_async_to_lds_b128)
#define ASYNC_LDS 1
#else
#define ASYNC_LDS 0
#endif

static __device__ inline void cp16_g2l(_Float16* lds_dst, const _Float16* gsrc) {
#if ASYNC_LDS
  __builtin_amdgcn_global_load_async_to_lds_b128(
      (__attribute__((address_space(1))) v4i*)gsrc,
      (__attribute__((address_space(3))) v4i*)lds_dst,
      /*offset=*/0, /*cpol=*/0);
#else
  *(uint4*)lds_dst = *(const uint4*)gsrc;
#endif
}

static __device__ inline void wait_async_copies() {
#if ASYNC_LDS
#if __has_builtin(__builtin_amdgcn_s_wait_asynccnt)
  __builtin_amdgcn_s_wait_asynccnt(0);
#else
  asm volatile("s_wait_asynccnt 0x0" ::: "memory");
#endif
#endif
}

// ---- WMMA wrapper (codegen-verified builtin, 8 args) ----------------------
static __device__ inline v8f wmma_f16(v16h a, v16h b, v8f c) {
  return __builtin_amdgcn_wmma_f32_16x16x32_f16(
      /*neg_a=*/false, a, /*neg_b=*/false, b,
      /*c_mod=*/(short)0, c, /*reuse_a=*/false, /*reuse_b=*/false);
}

// ---- fragment loads -------------------------------------------------------
// A (16x32 f16, row-major source, 32 halves per row):
//   lanes 0-15 : M=lane,    K = {0..7, 16..23}
//   lanes16-31 : M=lane-16, K = {8..15, 24..31}
static __device__ inline v16h ldfrag_a(const _Float16* prow, int lane) {
  union { v16h h; uint4 u[2]; } f;
  const _Float16* p = prow + ((lane & 16) ? 8 : 0);
  f.u[0] = *(const uint4*)(p);
  f.u[1] = *(const uint4*)(p + 16);
  return f.h;
}
// B (32x16 f16; source stored N-major so each lane reads its column contiguously):
//   lanes 0-15 : N=lane,    K = {0..7, 16..23}
//   lanes16-31 : N=lane-16, K = {8..15, 24..31}
static __device__ inline v16h ldfrag_b(const _Float16* prow, int lane) {
  union { v16h h; uint4 u[2]; } f;
  const _Float16* p = prow + ((lane & 16) ? 8 : 0);
  f.u[0] = *(const uint4*)(p);
  f.u[1] = *(const uint4*)(p + 16);
  return f.h;
}

// ---------------------------------------------------------------------------
// Double-buffered GEMM mainloop: workgroup tile 128(M) x 128(N), 8 waves in a
// 4(M) x 2(N) grid; each wave owns 32x64 = 8 WMMA accumulators. A row-major
// [M][K], Bt pre-transposed [N][K]. Per k-step: async-prefetch tile k+1 into
// the other LDS buffer while issuing 8 WMMAs from the current one; a single
// barrier per k-step both publishes the new tile and retires the old one.
// The k-loop is kept rolled so the accumulators keep one fixed register
// assignment (the x2 unroll was generating v_mov_b64 accumulator rotation +
// WMMA-hazard v_nops).
// ---------------------------------------------------------------------------
#define ATILE (128 * 32)
#define BTILE (128 * 32)

static __device__ inline void gemm_prefetch(const _Float16* __restrict__ A,
                                            const _Float16* __restrict__ Bt,
                                            int K, size_t m0, size_t n0, int k0,
                                            _Float16* Ab, _Float16* Bb, int tid) {
#pragma unroll
  for (int i = 0; i < 2; ++i) {           // 512 uint4 per 128x32 tile
    int u = tid + i * 256;
    int r = u >> 2, s4 = u & 3;
    cp16_g2l(Ab + r * 32 + s4 * 8, A + (m0 + (size_t)r) * K + k0 + s4 * 8);
    cp16_g2l(Bb + r * 32 + s4 * 8, Bt + (n0 + (size_t)r) * K + k0 + s4 * 8);
  }
}

static __device__ inline void gemm_tile(const _Float16* __restrict__ A,
                                        const _Float16* __restrict__ Bt,
                                        int K, size_t m0, size_t n0,
                                        _Float16* As, _Float16* Bs,
                                        v8f acc[8]) {
  const int tid  = threadIdx.x;
  const int lane = tid & 31;
  const int wave = tid >> 5;
  const int wm   = wave & 3;   // M strip (32 rows)
  const int wn   = wave >> 2;  // N group (64 cols)
  const int nk   = K >> 5;

  gemm_prefetch(A, Bt, K, m0, n0, 0, As, Bs, tid);
#pragma clang loop unroll(disable)
  for (int ks = 0; ks < nk; ++ks) {
    const int cur = ks & 1;
    wait_async_copies();
    __syncthreads();           // publish buf[cur]; retire readers of buf[1-cur]
    if (ks + 1 < nk)
      gemm_prefetch(A, Bt, K, m0, n0, (ks + 1) << 5,
                    As + ((ks + 1) & 1) * ATILE, Bs + ((ks + 1) & 1) * BTILE, tid);
    const _Float16* Ab = As + cur * ATILE;
    const _Float16* Bb = Bs + cur * BTILE;
    v16h af[2];
#pragma unroll
    for (int a2 = 0; a2 < 2; ++a2)
      af[a2] = ldfrag_a(Ab + (wm * 32 + a2 * 16 + (lane & 15)) * 32, lane);
#pragma unroll
    for (int t = 0; t < 4; ++t) {
      v16h bf = ldfrag_b(Bb + (wn * 64 + t * 16 + (lane & 15)) * 32, lane);
#pragma unroll
      for (int a2 = 0; a2 < 2; ++a2)
        acc[a2 * 4 + t] = wmma_f16(af[a2], bf, acc[a2 * 4 + t]);
    }
  }
}

// ---------------------------------------------------------------------------
// Stage 0: prep kernels
// ---------------------------------------------------------------------------
__global__ __launch_bounds__(256) void prep_x(const float* __restrict__ x,
                                              _Float16* __restrict__ xh) {
  size_t idx = (size_t)blockIdx.x * blockDim.x + threadIdx.x;
  if (idx >= (size_t)BATCH * NP * DIMC) return;
  int c = (int)(idx % DIMC);
  int r = (int)((idx / DIMC) % NP);
  int b = (int)(idx / ((size_t)DIMC * NP));
  float v = (r < NQ) ? x[((size_t)b * NQ + r) * DIMC + c] : 0.0f;
  xh[idx] = (_Float16)v;
}

// w[rows][cols] (f32) -> wT[cols][rows] (f16)
__global__ __launch_bounds__(256) void prep_wT(const float* __restrict__ w,
                                               _Float16* __restrict__ wT,
                                               int rows, int cols) {
  size_t idx = (size_t)blockIdx.x * blockDim.x + threadIdx.x;
  if (idx >= (size_t)rows * cols) return;
  int r = (int)(idx % rows);
  int c = (int)(idx / rows);
  wT[idx] = (_Float16)w[(size_t)r * cols + c];
}

__global__ __launch_bounds__(256) void prep_m(const float* __restrict__ dist_map,
                                              const int* __restrict__ dist_idx,
                                              float* __restrict__ Mf, int table) {
  size_t idx = (size_t)blockIdx.x * blockDim.x + threadIdx.x;
  if (idx >= (size_t)HEADS * NP * NP) return;
  int j = (int)(idx % NP);
  int i = (int)((idx / NP) % NP);
  int h = (int)(idx / ((size_t)NP * NP));
  float v = 0.0f;
  if (i < NQ && j < NQ) v = dist_map[(size_t)h * table + dist_idx[i * NQ + j]];
  Mf[idx] = v;
}

// ---------------------------------------------------------------------------
// Stage 1: QKV GEMM  [B*NP, 768] x [768, 2304] -> scatter Q (scaled), K, V^T
// ---------------------------------------------------------------------------
__global__ __launch_bounds__(256) void qkv_gemm(const _Float16* __restrict__ xh,
                                                const _Float16* __restrict__ wT,
                                                _Float16* __restrict__ qh,
                                                _Float16* __restrict__ kh,
                                                _Float16* __restrict__ vt,
                                                float scale) {
  __shared__ __align__(16) _Float16 As[2 * ATILE];
  __shared__ __align__(16) _Float16 Bs[2 * BTILE];
  size_t n0 = (size_t)blockIdx.x * 128;
  size_t m0 = (size_t)blockIdx.y * 128;
  v8f acc[8] = {{0}, {0}, {0}, {0}, {0}, {0}, {0}, {0}};
  gemm_tile(xh, wT, DIMC, m0, n0, As, Bs, acc);

  const int lane = threadIdx.x & 31, wave = threadIdx.x >> 5;
  const int wm = wave & 3, wn = wave >> 2;
  const int rl_hi = (lane >> 4) << 3;
#pragma unroll
  for (int a2 = 0; a2 < 2; ++a2) {
    const int rbase = (int)m0 + wm * 32 + a2 * 16 + rl_hi;
#pragma unroll
    for (int t = 0; t < 4; ++t) {
      int gc = (int)n0 + wn * 64 + t * 16 + (lane & 15);
      int which = gc / DIMC;
      int hh = (gc % DIMC) / HD;
      int d = gc & (HD - 1);
#pragma unroll
      for (int v = 0; v < 8; ++v) {
        int gm = rbase + v;
        int b = gm / NP, r = gm % NP;
        size_t bh = (size_t)(b * HEADS + hh);
        float val = acc[a2 * 4 + t][v];
        if (which == 0)      qh[(bh * NP + r) * HD + d] = (_Float16)(val * scale);
        else if (which == 1) kh[(bh * NP + r) * HD + d] = (_Float16)val;
        else                 vt[(bh * HD + d) * NP + r] = (_Float16)val; // V^T
      }
    }
  }
}

// ---------------------------------------------------------------------------
// Stage 2: fused attention per (b,h). K and V^T staged once in LDS (async);
// 8 waves, each owns a 16-row query strip (x2 iterations), full 16x224 score
// panel in f32 accumulators, bias*mask, butterfly softmax, LDS relayout, PV.
// Total LDS: 28KB (K) + 28KB (V^T) + 56KB (P) = 112KB of the 320KB WGP pool.
// ---------------------------------------------------------------------------
__global__ __launch_bounds__(256) void attn_kernel(const _Float16* __restrict__ qh,
                                                   const _Float16* __restrict__ kh,
                                                   const _Float16* __restrict__ vt,
                                                   const float* __restrict__ Mf,
                                                   _Float16* __restrict__ ah) {
  __shared__ __align__(16) _Float16 Ks[NP * HD];      // K   [224][64]
  __shared__ __align__(16) _Float16 Vs[HD * NP];      // V^T [64][224]
  __shared__ __align__(16) _Float16 P[8 * 16 * NP];   // per-wave P staging
  const int bh = blockIdx.x;               // = b*HEADS + h
  const int b = bh / HEADS, h = bh % HEADS;
  const _Float16* Q  = qh + (size_t)bh * NP * HD;
  const _Float16* Km = kh + (size_t)bh * NP * HD;
  const _Float16* V  = vt + (size_t)bh * HD * NP;
  const float* M     = Mf + (size_t)h * NP * NP;
  const int tid = threadIdx.x;
  const int lane = tid & 31, wave = tid >> 5;
  _Float16* Pw = P + wave * 16 * NP;
  const int colb  = lane & 15;
  const int rl_hi = (lane >> 4) << 3;

  // ---- stage K and V^T into LDS (async, 1792 x b128 each) -----------------
#pragma unroll
  for (int i = 0; i < 7; ++i) {
    int u = tid + i * 256;
    cp16_g2l(Ks + u * 8, Km + u * 8);
    cp16_g2l(Vs + u * 8, V + u * 8);
  }
  wait_async_copies();
  __syncthreads();

  for (int iter = 0; iter < 2; ++iter) {
    const int strip = wave + iter * 8;     // uniform per wave
    if (strip >= NP / 16) break;
    const int row0 = strip * 16;

    // ---- S = (Q*scale) @ K^T : 14 tiles of 16x16, K-dim = 64 (2 steps) ----
    v8f s[14];
#pragma unroll
    for (int j = 0; j < 14; ++j) s[j] = (v8f){0};
#pragma unroll
    for (int kk = 0; kk < HD; kk += 32) {
      v16h a = ldfrag_a(Q + (size_t)(row0 + colb) * HD + kk, lane);
#pragma unroll
      for (int j = 0; j < 14; ++j) {
        v16h bf = ldfrag_b(Ks + (j * 16 + colb) * HD + kk, lane);
        s[j] = wmma_f16(a, bf, s[j]);
      }
    }

    // ---- bias * mask + softmax (rows live across 16-lane halves) ----------
#pragma unroll
    for (int v = 0; v < 8; ++v) {
      const int row = row0 + v + rl_hi;
      float mx = -1e30f;
#pragma unroll
      for (int j = 0; j < 14; ++j) {
        int col = j * 16 + colb;
        float val = (col < NQ) ? s[j][v] * M[(size_t)row * NP + col] : -1e30f;
        s[j][v] = val;
        mx = fmaxf(mx, val);
      }
#pragma unroll
      for (int d = 1; d < 16; d <<= 1) mx = fmaxf(mx, __shfl_xor(mx, d, 32));
      float sum = 0.0f;
#pragma unroll
      for (int j = 0; j < 14; ++j) {
        float e = __expf(s[j][v] - mx);
        s[j][v] = e;
        sum += e;
      }
#pragma unroll
      for (int d = 1; d < 16; d <<= 1) sum += __shfl_xor(sum, d, 32);
      float inv = 1.0f / sum;
#pragma unroll
      for (int j = 0; j < 14; ++j)
        Pw[(v + rl_hi) * NP + j * 16 + colb] = (_Float16)(s[j][v] * inv);
    }

    // ---- O = P @ V : 4 tiles of 16x16, K-dim = 224 (7 steps) --------------
    v8f o[4] = {{0}, {0}, {0}, {0}};
#pragma unroll
    for (int kk = 0; kk < NP; kk += 32) {
      v16h a = ldfrag_a(Pw + colb * NP + kk, lane);
#pragma unroll
      for (int t = 0; t < 4; ++t) {
        v16h bf = ldfrag_b(Vs + (t * 16 + colb) * NP + kk, lane);
        o[t] = wmma_f16(a, bf, o[t]);
      }
    }
#pragma unroll
    for (int t = 0; t < 4; ++t) {
      int dcol = t * 16 + colb;
#pragma unroll
      for (int v = 0; v < 8; ++v) {
        int row = row0 + v + rl_hi;
        ah[((size_t)b * NP + row) * DIMC + h * HD + dcol] = (_Float16)o[t][v];
      }
    }
  }
}

// ---------------------------------------------------------------------------
// Stage 3: proj GEMM + bias, store only the 197 valid rows (fp32 out).
// ---------------------------------------------------------------------------
__global__ __launch_bounds__(256) void proj_gemm(const _Float16* __restrict__ ah,
                                                 const _Float16* __restrict__ wT,
                                                 const float* __restrict__ bias,
                                                 float* __restrict__ out) {
  __shared__ __align__(16) _Float16 As[2 * ATILE];
  __shared__ __align__(16) _Float16 Bs[2 * BTILE];
  size_t n0 = (size_t)blockIdx.x * 128;
  size_t m0 = (size_t)blockIdx.y * 128;
  v8f acc[8] = {{0}, {0}, {0}, {0}, {0}, {0}, {0}, {0}};
  gemm_tile(ah, wT, DIMC, m0, n0, As, Bs, acc);

  const int lane = threadIdx.x & 31, wave = threadIdx.x >> 5;
  const int wm = wave & 3, wn = wave >> 2;
  const int rl_hi = (lane >> 4) << 3;
#pragma unroll
  for (int a2 = 0; a2 < 2; ++a2) {
    const int rbase = (int)m0 + wm * 32 + a2 * 16 + rl_hi;
#pragma unroll
    for (int t = 0; t < 4; ++t) {
      int gc = (int)n0 + wn * 64 + t * 16 + (lane & 15);
      float bv = bias[gc];
#pragma unroll
      for (int v = 0; v < 8; ++v) {
        int gm = rbase + v;
        int b = gm / NP, r = gm % NP;
        if (r < NQ) out[((size_t)b * NQ + r) * DIMC + gc] = acc[a2 * 4 + t][v] + bv;
      }
    }
  }
}

// ---------------------------------------------------------------------------
extern "C" void kernel_launch(void* const* d_in, const int* in_sizes, int n_in,
                              void* d_out, int out_size, void* d_ws, size_t ws_size,
                              hipStream_t stream) {
  const float* x        = (const float*)d_in[0];
  const float* qkv_w    = (const float*)d_in[1];
  const float* proj_w   = (const float*)d_in[2];
  const float* proj_b   = (const float*)d_in[3];
  const float* dist_map = (const float*)d_in[4];
  const int*   dist_idx = (const int*)d_in[5];
  float* out = (float*)d_out;
  const int table = in_sizes[4] / HEADS; // 192

  // Workspace carve-up (all chunks are naturally 256B-aligned; total ~112 MB)
  char* ws = (char*)d_ws;
  size_t off = 0;
  auto take = [&](size_t bytes) { char* p = ws + off; off += bytes; return p; };
  _Float16* xh    = (_Float16*)take((size_t)BATCH * NP * DIMC * 2);
  _Float16* wqkvT = (_Float16*)take((size_t)3 * DIMC * DIMC * 2);
  _Float16* wprjT = (_Float16*)take((size_t)DIMC * DIMC * 2);
  _Float16* qh    = (_Float16*)take((size_t)BATCH * HEADS * NP * HD * 2);
  _Float16* kh    = (_Float16*)take((size_t)BATCH * HEADS * NP * HD * 2);
  _Float16* vt    = (_Float16*)take((size_t)BATCH * HEADS * HD * NP * 2);
  _Float16* ah    = (_Float16*)take((size_t)BATCH * NP * DIMC * 2);
  float*    Mf    = (float*)take((size_t)HEADS * NP * NP * 4);
  (void)ws_size; (void)n_in; (void)out_size;

  // stage 0
  {
    size_t n = (size_t)BATCH * NP * DIMC;
    prep_x<<<dim3((unsigned)((n + 255) / 256)), dim3(256), 0, stream>>>(x, xh);
  }
  {
    size_t n = (size_t)DIMC * 3 * DIMC;
    prep_wT<<<dim3((unsigned)((n + 255) / 256)), dim3(256), 0, stream>>>(
        qkv_w, wqkvT, DIMC, 3 * DIMC);
  }
  {
    size_t n = (size_t)DIMC * DIMC;
    prep_wT<<<dim3((unsigned)((n + 255) / 256)), dim3(256), 0, stream>>>(
        proj_w, wprjT, DIMC, DIMC);
  }
  {
    size_t n = (size_t)HEADS * NP * NP;
    prep_m<<<dim3((unsigned)((n + 255) / 256)), dim3(256), 0, stream>>>(
        dist_map, dist_idx, Mf, table);
  }

  // stage 1: QKV GEMM. M = 64*224 = 14336, N = 2304, K = 768.
  qkv_gemm<<<dim3(3 * DIMC / 128, BATCH * NP / 128), dim3(256), 0, stream>>>(
      xh, wqkvT, qh, kh, vt, 0.125f /* HD^-0.5 folded into Q */);

  // stage 2: fused attention, one workgroup per (batch, head)
  attn_kernel<<<dim3(BATCH * HEADS), dim3(256), 0, stream>>>(qh, kh, vt, Mf, ah);

  // stage 3: proj GEMM + bias. M = 14336, N = 768, K = 768.
  proj_gemm<<<dim3(DIMC / 128, BATCH * NP / 128), dim3(256), 0, stream>>>(
      ah, wprjT, proj_b, out);
}
